// SetAbstractionMSG_28097676051191
// MI455X (gfx1250) — matrix operand order repaired and checked by reading, not compiled, not run on hardware
//
#include <hip/hip_runtime.h>
#include <hip/hip_bf16.h>

#define BB   4
#define NN   8192
#define CIN  64
#define NC   1024
#define EPSV 1e-5f

typedef __attribute__((ext_vector_type(16))) _Float16 v16h;
typedef __attribute__((ext_vector_type(8)))  float    v8f;

struct LayerP { const float* W; const float* gamma; const float* beta; const float* mean; const float* var; };
struct MlpP   { LayerP l[3]; };

// ---------------------------------------------------------------------------
// Kernel 1: farthest point sampling, one 1024-thread block per batch.
// mind[] lives in LDS; each step: block argmax (lowest index wins ties, like
// jnp.argmax) then min-distance update against the newly selected point.
// ---------------------------------------------------------------------------
__global__ __launch_bounds__(1024) void fps_kernel(const float* __restrict__ point,
                                                   float* __restrict__ cent_ws,
                                                   float* __restrict__ cent_out)
{
    __shared__ float mind[NN];
    __shared__ float redv[1024];
    __shared__ int   redi[1024];
    __shared__ float cur[3];

    const int b = blockIdx.x;
    const int t = threadIdx.x;
    const float* pts = point + (size_t)b * NN * 3;

    const float p0x = pts[0], p0y = pts[1], p0z = pts[2];
    for (int i = t; i < NN; i += 1024) {
        float dx = pts[3*i+0] - p0x, dy = pts[3*i+1] - p0y, dz = pts[3*i+2] - p0z;
        mind[i] = dx*dx + dy*dy + dz*dz;
    }
    if (t == 0) {
        size_t o = (size_t)b * NC * 3;
        cent_ws[o+0] = p0x; cent_ws[o+1] = p0y; cent_ws[o+2] = p0z;
        cent_out[o+0] = p0x; cent_out[o+1] = p0y; cent_out[o+2] = p0z;
    }
    __syncthreads();

    for (int s = 1; s < NC; ++s) {
        float bv = -1.0f; int bi = NN;
        for (int i = t; i < NN; i += 1024) {
            float v = mind[i];
            if (v > bv) { bv = v; bi = i; }
        }
        redv[t] = bv; redi[t] = bi;
        __syncthreads();
        for (int off = 512; off > 0; off >>= 1) {
            if (t < off) {
                float ov = redv[t+off]; int oi = redi[t+off];
                if (ov > redv[t] || (ov == redv[t] && oi < redi[t])) { redv[t] = ov; redi[t] = oi; }
            }
            __syncthreads();
        }
        if (t == 0) {
            int sel = redi[0];
            float x = pts[3*sel+0], y = pts[3*sel+1], z = pts[3*sel+2];
            cur[0] = x; cur[1] = y; cur[2] = z;
            size_t o = ((size_t)b * NC + s) * 3;
            cent_ws[o+0] = x; cent_ws[o+1] = y; cent_ws[o+2] = z;
            cent_out[o+0] = x; cent_out[o+1] = y; cent_out[o+2] = z;
        }
        __syncthreads();
        const float cx = cur[0], cy = cur[1], cz = cur[2];
        for (int i = t; i < NN; i += 1024) {
            float dx = pts[3*i+0] - cx, dy = pts[3*i+1] - cy, dz = pts[3*i+2] - cz;
            float d = dx*dx + dy*dy + dz*dz;
            float m = mind[i];
            mind[i] = d < m ? d : m;
        }
        __syncthreads();
    }
}

// ---------------------------------------------------------------------------
// Kernel 2: ball query, one wave32 per (batch, centroid), all three radii in
// one pass. Ordered writes via ballot + prefix popcount; early exit when full.
// ---------------------------------------------------------------------------
__global__ __launch_bounds__(256) void ball_query_kernel(const float* __restrict__ point,
                                                         const float* __restrict__ cent,
                                                         int* __restrict__ idx0,
                                                         int* __restrict__ idx1,
                                                         int* __restrict__ idx2)
{
    const int wave = (blockIdx.x * blockDim.x + threadIdx.x) >> 5;
    const int lane = threadIdx.x & 31;
    if (wave >= BB * NC) return;

    const int b = wave / NC;
    const float* pts = point + (size_t)b * NN * 3;
    const float cx = cent[wave*3+0], cy = cent[wave*3+1], cz = cent[wave*3+2];

    int* o0 = idx0 + (size_t)wave * 16;
    int* o1 = idx1 + (size_t)wave * 32;
    int* o2 = idx2 + (size_t)wave * 64;
    if (lane < 16) o0[lane] = -1;
    o1[lane] = -1;
    o2[lane] = -1; o2[lane + 32] = -1;

    const float r0 = 0.1f*0.1f, r1 = 0.2f*0.2f, r2 = 0.4f*0.4f;
    int c0 = 0, c1 = 0, c2 = 0;
    const unsigned ltm = (1u << lane) - 1u;

    for (int base = 0; base < NN; base += 32) {
        int i = base + lane;
        float dx = pts[3*i+0] - cx, dy = pts[3*i+1] - cy, dz = pts[3*i+2] - cz;
        float d2 = dx*dx + dy*dy + dz*dz;
        bool in0 = d2 < r0, in1 = d2 < r1, in2 = d2 < r2;
        unsigned m0 = (unsigned)__ballot(in0);
        unsigned m1 = (unsigned)__ballot(in1);
        unsigned m2 = (unsigned)__ballot(in2);
        if (in0) { int p = c0 + __popc(m0 & ltm); if (p < 16) o0[p] = i; }
        if (in1) { int p = c1 + __popc(m1 & ltm); if (p < 32) o1[p] = i; }
        if (in2) { int p = c2 + __popc(m2 & ltm); if (p < 64) o2[p] = i; }
        c0 += __popc(m0); c1 += __popc(m1); c2 += __popc(m2);
        if (c0 >= 16 && c1 >= 32 && c2 >= 64) break;
    }
}

// ---------------------------------------------------------------------------
// Kernel 3: grouped MLP per branch. One block (8 wave32) per centroid.
// WMMA f16->f32, 16x16x32 tiles. A-frag: 16-bit A 16x32 layout. B-frag: W^T
// per 16-bit B 32x16 layout. BN+ReLU applied on the f32 C/D fragment.
// ---------------------------------------------------------------------------
template<int KN, int O1, int O2, int O3>
__global__ __launch_bounds__(256) void mlp_kernel(const float* __restrict__ point,
                                                  const float* __restrict__ feat,
                                                  const float* __restrict__ cent,
                                                  const int*   __restrict__ idx,
                                                  MlpP mp,
                                                  float* __restrict__ outFeat,
                                                  int outOff)
{
    constexpr int LD = 136;                 // padded act row stride (f16) -> no bank conflicts
    constexpr int W0 = O1 * (96 + 8);
    constexpr int W1 = O2 * (O1 + 8);
    constexpr int W2 = O3 * (O2 + 8);
    constexpr int WMAX = (W0 > W1) ? (W0 > W2 ? W0 : W2) : (W1 > W2 ? W1 : W2);

    __shared__ _Float16 Wl[WMAX];
    __shared__ _Float16 Abuf[2][KN * LD];
    __shared__ float bnsc[128];
    __shared__ float bnsh[128];

    const int bid = blockIdx.x;             // b*NC + n
    const int b   = bid / NC;
    const int t   = threadIdx.x;

    const float cx = cent[bid*3+0], cy = cent[bid*3+1], cz = cent[bid*3+2];
    const int*   nb  = idx  + (size_t)bid * KN;
    const float* pts = point + (size_t)b * NN * 3;
    const float* fts = feat  + (size_t)b * NN * CIN;

    // ---- gather [KN, 67] -> f16 LDS, padded to LD (masked_gather semantics)
    for (int e = t; e < KN * LD; e += 256) {
        int k = e / LD, c = e % LD;
        int j = nb[k];
        float v = 0.0f;
        if (c < CIN)           v = (j >= 0) ? fts[(size_t)j*CIN + c]  : 0.0f;
        else if (c == CIN)     v = (j >= 0) ? pts[3*j+0] - cx : -cx;
        else if (c == CIN + 1) v = (j >= 0) ? pts[3*j+1] - cy : -cy;
        else if (c == CIN + 2) v = (j >= 0) ? pts[3*j+2] - cz : -cz;
        Abuf[0][k*LD + c] = (_Float16)v;
    }

    constexpr int kinA[3] = {67, O1, O2};   // actual input channels
    constexpr int kdA [3] = {96, O1, O2};   // padded K dim (mult of 32)
    constexpr int kwpA[3] = {96 + 8, O1 + 8, O2 + 8}; // padded W row stride
    constexpr int ocA [3] = {O1, O2, O3};

    int buf = 0;
    #pragma unroll
    for (int l = 0; l < 3; ++l) {
        const LayerP& L = mp.l[l];
        const int kia = kinA[l], kda = kdA[l], kwp = kwpA[l], oca = ocA[l];

        __syncthreads();                    // prior layer reads done; gather visible
        for (int e = t; e < oca * kwp; e += 256) {
            int o = e / kwp, c = e % kwp;
            Wl[e] = (_Float16)((c < kia) ? L.W[o*kia + c] : 0.0f);
        }
        if (t < oca) {
            float sc = L.gamma[t] * rsqrtf(L.var[t] + EPSV);
            bnsc[t] = sc;
            bnsh[t] = L.beta[t] - L.mean[t] * sc;
        }
        __syncthreads();

        const int tilesN = oca / 16;
        const int tot    = (KN / 16) * tilesN;
        const int wid  = t >> 5;
        const int lane = t & 31;
        const int h    = lane >> 4;
        const int lm   = lane & 15;

        for (int tile = wid; tile < tot; tile += 8) {
            const int tm = tile / tilesN, tn = tile % tilesN;
            v8f acc = {};
            for (int kb = 0; kb < kda; kb += 32) {
                v16h a, w;
                const int m = tm*16 + lm;
                #pragma unroll
                for (int j = 0; j < 16; ++j) {          // A 16x32 f16 layout
                    int p  = j >> 1;
                    int kk = kb + ((p & 3) * 2) + ((p >> 2) * 16) + h*8 + (j & 1);
                    a[j] = Abuf[buf][m*LD + kk];
                }
                const int n = tn*16 + lm;
                #pragma unroll
                for (int j = 0; j < 16; ++j) {          // B 32x16 f16 layout (W^T)
                    int kk = kb + h*8 + (j & 7) + (j >> 3) * 16;
                    w[j] = Wl[n*kwp + kk];
                }
                acc = __builtin_amdgcn_wmma_f32_16x16x32_f16(
                          false, a, false, w, (short)0, acc, false, false);
            }
            const int n = tn*16 + lm;
            const float sc = bnsc[n], sh = bnsh[n];
            #pragma unroll
            for (int v = 0; v < 8; ++v) {               // C/D: M = v + h*8
                int m = tm*16 + v + h*8;
                float y = acc[v]*sc + sh;
                y = y > 0.0f ? y : 0.0f;
                Abuf[buf ^ 1][m*LD + n] = (_Float16)y;
            }
        }
        buf ^= 1;
    }

    __syncthreads();
    if (t < O3) {                                       // max over K neighbors (ReLU -> >= 0)
        float mx = 0.0f;
        for (int m = 0; m < KN; ++m) {
            float v = (float)Abuf[buf][m*LD + t];
            mx = v > mx ? v : mx;
        }
        outFeat[(size_t)bid * 384 + outOff + t] = mx;
    }
}

// ---------------------------------------------------------------------------
extern "C" void kernel_launch(void* const* d_in, const int* in_sizes, int n_in,
                              void* d_out, int out_size, void* d_ws, size_t ws_size,
                              hipStream_t stream)
{
    const float* point = (const float*)d_in[0];
    const float* feat  = (const float*)d_in[1];

    auto LP = [&](int br, int ly) -> LayerP {
        int base = 2 + (br * 3 + ly) * 5;   // W, gamma, beta, mean, var
        return LayerP{ (const float*)d_in[base+0], (const float*)d_in[base+1],
                       (const float*)d_in[base+2], (const float*)d_in[base+3],
                       (const float*)d_in[base+4] };
    };
    MlpP m0{{ LP(0,0), LP(0,1), LP(0,2) }};
    MlpP m1{{ LP(1,0), LP(1,1), LP(1,2) }};
    MlpP m2{{ LP(2,0), LP(2,1), LP(2,2) }};

    float* outCent = (float*)d_out;                      // [B, NC, 3]
    float* outFeat = outCent + (size_t)BB * NC * 3;      // [B, NC, 384]

    float* cent = (float*)d_ws;                          // ws: centroids
    int* idx0 = (int*)(cent + (size_t)BB * NC * 3);      // [B*NC, 16]
    int* idx1 = idx0 + (size_t)BB * NC * 16;             // [B*NC, 32]
    int* idx2 = idx1 + (size_t)BB * NC * 32;             // [B*NC, 64]

    fps_kernel<<<BB, 1024, 0, stream>>>(point, cent, outCent);
    ball_query_kernel<<<(BB * NC * 32) / 256, 256, 0, stream>>>(point, cent, idx0, idx1, idx2);
    mlp_kernel<16, 64, 64, 128><<<BB * NC, 256, 0, stream>>>(point, feat, cent, idx0, m0, outFeat, 0);
    mlp_kernel<32, 64, 96, 128><<<BB * NC, 256, 0, stream>>>(point, feat, cent, idx1, m1, outFeat, 128);
    mlp_kernel<64, 96, 96, 128><<<BB * NC, 256, 0, stream>>>(point, feat, cent, idx2, m2, outFeat, 256);
}
